// TutelOutputOnlyMoE_51462298141174
// MI455X (gfx1250) — matrix operand order repaired and checked by reading, not compiled, not run on hardware
//
#include <hip/hip_runtime.h>

// ---------------------------------------------------------------------------
// Tutel top-1 MoE for MI455X (gfx1250, wave32).
// FFN = 137 GFLOP vs ~350MB traffic -> compute bound -> bf16 WMMA
// (v_wmma_f32_16x16x32_bf16, f32 accum). Gating in exact fp32.
// GEMMs: 128x128 tile, K-step 32, double-buffered LDS; A staged via
// GLOBAL_LOAD_ASYNC_TO_LDS_B128 (ASYNCcnt), B staged fp32->bf16 packed.
// ---------------------------------------------------------------------------

typedef __bf16 bf16;
typedef __bf16 v16bf __attribute__((ext_vector_type(16)));
typedef __bf16 v8bf  __attribute__((ext_vector_type(8)));
typedef __bf16 v2bf  __attribute__((ext_vector_type(2)));
typedef float  v8f   __attribute__((ext_vector_type(8)));

#define N_TOK 8192
#define MDIM  1024
#define HDIM  4096
#define NEXP  8
#define CAP   1024   // (N + E - 1) / E
#define BPAD  40     // LDS row pitch (elems): 80B -> 16 distinct banks across lanes

__device__ __forceinline__ void wait_async0() {
#if __has_builtin(__builtin_amdgcn_s_wait_asynccnt)
  __builtin_amdgcn_s_wait_asynccnt(0);
#else
  asm volatile("s_wait_asynccnt 0x0" ::: "memory");
#endif
}

// ---------------------------------------------------------------------------
// 0) init: zero d_out (dropped tokens stay 0), slot_tok = -1
// ---------------------------------------------------------------------------
__global__ __launch_bounds__(256) void init_kernel(float* __restrict__ out,
                                                   int* __restrict__ slot_tok) {
  int i = blockIdx.x * 256 + threadIdx.x;
  int stride = gridDim.x * 256;
  float4 z = {0.f, 0.f, 0.f, 0.f};
  float4* o4 = (float4*)out;
  for (int p = i; p < (N_TOK * MDIM) / 4; p += stride) o4[p] = z;
  if (i < NEXP * CAP) slot_tok[i] = -1;
}

// ---------------------------------------------------------------------------
// 1) gating: one wave per token, fp32 dots over M=1024 for E=8 experts.
// ---------------------------------------------------------------------------
__global__ __launch_bounds__(256) void gate_kernel(const float* __restrict__ x,
                                                   const float* __restrict__ wg,
                                                   int* __restrict__ tok_exp,
                                                   float* __restrict__ tok_gate) {
  const int wid  = threadIdx.x >> 5;
  const int lane = threadIdx.x & 31;
  const int n = blockIdx.x * 8 + wid;

  const float* xr = x + (size_t)n * MDIM + lane * 32;
  float acc[NEXP];
#pragma unroll
  for (int e = 0; e < NEXP; e++) acc[e] = 0.0f;

#pragma unroll
  for (int m = 0; m < 32; m++) {
    float xv = xr[m];
    const float* wr = wg + (size_t)(lane * 32 + m) * NEXP;
#pragma unroll
    for (int e = 0; e < NEXP; e++) acc[e] = fmaf(xv, wr[e], acc[e]);
  }
#pragma unroll
  for (int off = 16; off >= 1; off >>= 1) {
#pragma unroll
    for (int e = 0; e < NEXP; e++) acc[e] += __shfl_xor(acc[e], off, 32);
  }
  if (lane == 0) {
    float mx = acc[0];
    int bi = 0;
#pragma unroll
    for (int e = 1; e < NEXP; e++) {
      if (acc[e] > mx) { mx = acc[e]; bi = e; }
    }
    float s = 0.0f;
#pragma unroll
    for (int e = 0; e < NEXP; e++) s += __expf(acc[e] - mx);
    tok_exp[n]  = bi;
    tok_gate[n] = 1.0f / s;   // softmax prob of argmax expert
  }
}

// ---------------------------------------------------------------------------
// 2) capacity scan: token-order priority, single 1024-thread workgroup.
// ---------------------------------------------------------------------------
__global__ __launch_bounds__(1024) void scan_kernel(const int* __restrict__ tok_exp,
                                                    int* __restrict__ tok_dest,
                                                    int* __restrict__ slot_tok) {
  __shared__ int wcnt[32][NEXP];
  __shared__ int base[NEXP];
  const int tid = threadIdx.x;
  const int lane = tid & 31, wid = tid >> 5;

  if (tid < NEXP) base[tid] = 0;
  __syncthreads();

  for (int chunk = 0; chunk < N_TOK; chunk += 1024) {
    int t = chunk + tid;
    int e = tok_exp[t];

    unsigned bm[NEXP];
#pragma unroll
    for (int q = 0; q < NEXP; q++) bm[q] = (unsigned)__ballot(e == q);

    int lanerank = __popc(bm[e] & ((1u << lane) - 1u));
    if (lane < NEXP) wcnt[wid][lane] = __popc(bm[lane]);
    __syncthreads();

    int prefix = 0;
    for (int w = 0; w < wid; w++) prefix += wcnt[w][e];
    int loc = base[e] + prefix + lanerank;

    int dest = -1;
    if (loc < CAP) {
      dest = e * CAP + loc;
      slot_tok[dest] = t;
    }
    tok_dest[t] = dest;
    __syncthreads();

    if (tid < NEXP) {
      int tot = 0;
      for (int w = 0; w < 32; w++) tot += wcnt[w][tid];
      base[tid] += tot;
    }
    __syncthreads();
  }
}

// ---------------------------------------------------------------------------
// 3) dispatch: scatter kept token rows into bf16 Xd[E*CAP, M]
// ---------------------------------------------------------------------------
struct alignas(8) bf16x4 { bf16 a, b, c, d; };

__global__ __launch_bounds__(256) void dispatch_kernel(const float* __restrict__ x,
                                                       const int* __restrict__ tok_dest,
                                                       bf16* __restrict__ Xd) {
  int n = blockIdx.x;
  int dest = tok_dest[n];
  if (dest < 0) return;
  const float4* src = (const float4*)(x + (size_t)n * MDIM);
  float4 v = src[threadIdx.x];
  bf16x4 o = { (bf16)v.x, (bf16)v.y, (bf16)v.z, (bf16)v.w };
  ((bf16x4*)(Xd + (size_t)dest * MDIM))[threadIdx.x] = o;
}

// ---------------------------------------------------------------------------
// 4/5) grouped GEMM: D = A(bf16)[E][CAP][K] @ B(fp32->bf16)[E][K][NC]
//  MODE 0: +b1, relu, store bf16 Hb ; MODE 1: +b2, *gate, scatter to out.
//  Double-buffered LDS; A via async-to-LDS, B via packed cvt + b32 DS stores.
// ---------------------------------------------------------------------------
template <int MODE>
__global__ __launch_bounds__(256) void moe_gemm(const bf16*  __restrict__ Ag,
                                                const float* __restrict__ Bg,
                                                const float* __restrict__ bias,
                                                bf16*  __restrict__ Hb,
                                                float* __restrict__ out,
                                                const int*   __restrict__ slot_tok,
                                                const float* __restrict__ tok_gate,
                                                int K, int NC) {
  __shared__ bf16 As[2][128][BPAD];   // [buf][row][k]
  __shared__ bf16 Bs[2][128][BPAD];   // [buf][col_n][k] (transposed)

  const int e       = blockIdx.z;
  const int rowBase = blockIdx.y * 128;
  const int colBase = blockIdx.x * 128;
  const int tid  = threadIdx.x;
  const int lane = tid & 31, wid = tid >> 5;
  const int waveRow = (wid & 1) * 64;     // 2 row groups of 64
  const int waveCol = (wid >> 1) * 32;    // 4 col groups of 32
  const int l15   = lane & 15;
  const int khalf = lane >> 4;

  const bf16*  Ae = Ag + ((size_t)e * CAP + rowBase) * K;
  const float* Be = Bg + (size_t)e * K * NC + colBase;

  // A staging mapping: each thread async-copies 32B of one row
  const int arow = tid >> 1, ahalf = tid & 1;
  // B staging mapping: each thread owns a K-row pair x 8 cols
  const int kp   = tid >> 4;            // 0..15 -> k rows 2kp, 2kp+1
  const int cseg = (tid & 15) * 8;      // 0..120

  v8f acc[4][2];
#pragma unroll
  for (int i = 0; i < 4; i++)
#pragma unroll
    for (int j = 0; j < 2; j++)
#pragma unroll
      for (int q = 0; q < 8; q++) acc[i][j][q] = 0.0f;

  float r0[8], r1[8];

  auto stageA_async = [&](int k0, int buf) {
    const bf16* src = Ae + (size_t)arow * K + k0 + ahalf * 16;
    unsigned lds0 = (unsigned)(uintptr_t)&As[buf][arow][ahalf * 16];
    // INST_OFFSET applies to both LDS and global address (ISA 10.x async):
    asm volatile("global_load_async_to_lds_b128 %0, %1, off\n\t"
                 "global_load_async_to_lds_b128 %0, %1, off offset:16"
                 :: "v"(lds0), "v"(src)
                 : "memory");
  };
  auto loadB = [&](int k0) {
    const float* s0 = Be + (size_t)(k0 + 2 * kp) * NC + cseg;
    const float* s1 = s0 + NC;
    float4 a0 = *(const float4*)(s0);
    float4 a1 = *(const float4*)(s0 + 4);
    float4 b0 = *(const float4*)(s1);
    float4 b1 = *(const float4*)(s1 + 4);
    r0[0]=a0.x; r0[1]=a0.y; r0[2]=a0.z; r0[3]=a0.w;
    r0[4]=a1.x; r0[5]=a1.y; r0[6]=a1.z; r0[7]=a1.w;
    r1[0]=b0.x; r1[1]=b0.y; r1[2]=b0.z; r1[3]=b0.w;
    r1[4]=b1.x; r1[5]=b1.y; r1[6]=b1.z; r1[7]=b1.w;
  };
  auto storeB = [&](int buf) {
#pragma unroll
    for (int j = 0; j < 8; j++) {
      v2bf p;
      p[0] = (bf16)r0[j];
      p[1] = (bf16)r1[j];
      *(v2bf*)&Bs[buf][cseg + j][2 * kp] = p;   // 4B store, k-pair packed
    }
  };

  // ---- prologue: stage tile 0 into buffer 0 ----
  stageA_async(0, 0);
  loadB(0);
  storeB(0);
  wait_async0();
  __syncthreads();

  int buf = 0;
  for (int k0 = 0; k0 < K; k0 += 32, buf ^= 1) {
    const int nxt = buf ^ 1;
    const bool more = (k0 + 32) < K;
    if (more) {
      stageA_async(k0 + 32, nxt);   // async copy overlaps WMMA below
      loadB(k0 + 32);               // global loads issue early
    }

    // ---- fragments per ISA 16-bit A/B layouts ----
    v16bf a[4], b[2];
#pragma unroll
    for (int i = 0; i < 4; i++) {
      v8bf lo = *(const v8bf*)&As[buf][waveRow + i * 16 + l15][khalf * 8];
      v8bf hi = *(const v8bf*)&As[buf][waveRow + i * 16 + l15][16 + khalf * 8];
#pragma unroll
      for (int q = 0; q < 8; q++) { a[i][q] = lo[q]; a[i][8 + q] = hi[q]; }
    }
#pragma unroll
    for (int j = 0; j < 2; j++) {
      v8bf lo = *(const v8bf*)&Bs[buf][waveCol + j * 16 + l15][khalf * 8];
      v8bf hi = *(const v8bf*)&Bs[buf][waveCol + j * 16 + l15][16 + khalf * 8];
#pragma unroll
      for (int q = 0; q < 8; q++) { b[j][q] = lo[q]; b[j][8 + q] = hi[q]; }
    }

#pragma unroll
    for (int i = 0; i < 4; i++)
#pragma unroll
      for (int j = 0; j < 2; j++)
        acc[i][j] = __builtin_amdgcn_wmma_f32_16x16x32_bf16(
            false, a[i], false, b[j], (short)0, acc[i][j], false, false);

    if (more) {
      storeB(nxt);     // fill next B buffer
      wait_async0();   // next A buffer resident before barrier release
    }
    __syncthreads();
  }

  // ---- epilogue: C/D layout — VGPR r: (M = r or r+8 by lane[4], N = lane[3:0])
  const int rhi = khalf * 8;
#pragma unroll
  for (int i = 0; i < 4; i++) {
#pragma unroll
    for (int j = 0; j < 2; j++) {
      int col = colBase + waveCol + j * 16 + l15;
#pragma unroll
      for (int r = 0; r < 8; r++) {
        int row = rowBase + waveRow + i * 16 + rhi + r;   // row within expert
        float v = acc[i][j][r];
        if (MODE == 0) {
          v += bias[e * NC + col];
          v = v > 0.0f ? v : 0.0f;
          Hb[((size_t)e * CAP + row) * NC + col] = (bf16)v;
        } else {
          int t = slot_tok[e * CAP + row];
          if (t >= 0) {
            float g = tok_gate[t];
            out[(size_t)t * NC + col] = (v + bias[e * NC + col]) * g;
          }
        }
      }
    }
  }
}

// ---------------------------------------------------------------------------
// launcher
// ---------------------------------------------------------------------------
extern "C" void kernel_launch(void* const* d_in, const int* in_sizes, int n_in,
                              void* d_out, int out_size, void* d_ws, size_t ws_size,
                              hipStream_t stream) {
  (void)in_sizes; (void)n_in; (void)out_size; (void)ws_size;
  const float* x  = (const float*)d_in[0];
  const float* wg = (const float*)d_in[1];
  const float* w1 = (const float*)d_in[2];
  const float* b1 = (const float*)d_in[3];
  const float* w2 = (const float*)d_in[4];
  const float* b2 = (const float*)d_in[5];
  float* out = (float*)d_out;

  char* ws = (char*)d_ws;
  size_t off = 0;
  auto alloc = [&](size_t bytes) -> void* {
    void* p = ws + off;
    off += (bytes + 255) & ~(size_t)255;
    return p;
  };
  int*   tok_exp  = (int*)  alloc((size_t)N_TOK * 4);
  float* tok_gate = (float*)alloc((size_t)N_TOK * 4);
  int*   tok_dest = (int*)  alloc((size_t)N_TOK * 4);
  int*   slot_tok = (int*)  alloc((size_t)NEXP * CAP * 4);
  bf16*  Xd       = (bf16*) alloc((size_t)N_TOK * MDIM * 2);
  bf16*  Hb       = (bf16*) alloc((size_t)NEXP * CAP * HDIM * 2);

  init_kernel<<<dim3(2048), dim3(256), 0, stream>>>(out, slot_tok);
  gate_kernel<<<dim3(N_TOK / 8), dim3(256), 0, stream>>>(x, wg, tok_exp, tok_gate);
  scan_kernel<<<dim3(1), dim3(1024), 0, stream>>>(tok_exp, tok_dest, slot_tok);
  dispatch_kernel<<<dim3(N_TOK), dim3(256), 0, stream>>>(x, tok_dest, Xd);

  // GEMM1: Xd[E,CAP,M] @ w1[E,M,H] -> relu -> Hb[E,CAP,H] (bf16)
  moe_gemm<0><<<dim3(HDIM / 128, CAP / 128, NEXP), dim3(256), 0, stream>>>(
      Xd, w1, b1, Hb, nullptr, nullptr, nullptr, MDIM, HDIM);
  // GEMM2: Hb[E,CAP,H] @ w2[E,H,M] -> +b2, *gate, scatter to out
  moe_gemm<1><<<dim3(MDIM / 128, CAP / 128, NEXP), dim3(256), 0, stream>>>(
      Hb, w2, b2, nullptr, out, slot_tok, tok_gate, HDIM, MDIM);
}
